// ConvPool_77962246357654
// MI455X (gfx1250) — matrix operand order repaired and checked by reading, not compiled, not run on hardware
//
#include <hip/hip_runtime.h>
#include <hip/hip_bf16.h>

// ---------------------------------------------------------------------------
// CDNA5 (gfx1250) implementation: bf16 WMMA implicit-GEMM conv tower.
//   - intermediates ping-pong in bf16 through d_ws (L2-resident: 2x67MB < 192MB)
//   - weights pre-packed into the 16x32 bf16 A-fragment lane layout
//   - activations position-major; B slabs double-buffered in LDS via
//     GLOBAL_LOAD_ASYNC_TO_LDS_B128 (ASYNCcnt) when the toolchain has it
// ---------------------------------------------------------------------------

typedef __attribute__((ext_vector_type(16))) __bf16 v16bf;
typedef __attribute__((ext_vector_type(8)))  float  v8f;
typedef int v4i_vec __attribute__((vector_size(16)));

union BFrag {
    v16bf v;
    uint4 u[2];
    __bf16 e[16];
};

#define HID    256
#define LC     512
#define NCHUNK 256          // B(2) * max_n_peaks(128)
#define MOTIF  283
#define KPROJ  288          // 283 padded to multiple of 32

#if defined(__has_builtin)
#if __has_builtin(__builtin_amdgcn_global_load_async_to_lds_b128)
#define USE_ASYNC_LDS 1
#endif
#endif

__device__ __forceinline__ void wait_async_all() {
#if defined(USE_ASYNC_LDS)
#if __has_builtin(__builtin_amdgcn_s_wait_asynccnt)
    __builtin_amdgcn_s_wait_asynccnt(0);
#else
    asm volatile("s_wait_asynccnt 0" ::: "memory");
#endif
#endif
}

// ---------------------------------------------------------------------------
// Pack w_proj (H=256 x M=283) into A-fragment layout [mt(16)][kt(9)][lane(32)][j(16)]
// A 16x32 bf16 layout used here (contiguous k, same as B): lane<16 -> K {0..15}, lane>=16 -> K {16..31}
// ---------------------------------------------------------------------------
__global__ void pack_proj_kernel(const float* __restrict__ w, __bf16* __restrict__ dst) {
    int idx = blockIdx.x * 256 + threadIdx.x;
    if (idx >= 16 * 9 * 32 * 16) return;
    int j    = idx & 15;
    int lane = (idx >> 4) & 31;
    int kt   = (idx >> 9) % 9;
    int mt   = (idx >> 9) / 9;
    int m    = mt * 16 + (lane & 15);
    int khalf = lane >> 4;
    int j2 = j >> 1, p = j & 1;
    int kk = 16 * khalf + 2 * j2 + p;
    int k  = kt * 32 + kk;
    float v = (k < MOTIF) ? w[m * MOTIF + k] : 0.0f;
    dst[idx] = (__bf16)v;
}

// ---------------------------------------------------------------------------
// Pack tower_w (7, O=256, I=256, 3) into [layer][mt(16)][kt(24)][lane(32)][j(16)]
// K ordered as k = tap*256 + ci  (so each 32-wide K tile has a single tap)
// ---------------------------------------------------------------------------
__global__ void pack_tower_kernel(const float* __restrict__ w, __bf16* __restrict__ dst) {
    int idx = blockIdx.x * 256 + threadIdx.x;
    if (idx >= 7 * 16 * 24 * 32 * 16) return;
    int j    = idx & 15;
    int lane = (idx >> 4) & 31;
    int rem  = idx >> 9;           // layer*384 + mt*24 + kt
    int kt   = rem % 24; rem /= 24;
    int mt   = rem % 16;
    int layer = rem / 16;
    int m    = mt * 16 + (lane & 15);
    int khalf = lane >> 4;
    int j2 = j >> 1, p = j & 1;
    int kk = 16 * khalf + 2 * j2 + p;
    int k  = kt * 32 + kk;
    int tap = k >> 8;
    int ci  = k & 255;
    dst[idx] = (__bf16)w[(((size_t)layer * HID + m) * HID + ci) * 3 + tap];
}

// ---------------------------------------------------------------------------
// Projection GEMM: Y[pos][o] = sum_k x[pos][k] * Wproj[o][k], bf16 WMMA, f32 acc.
// Grid: (8 n-blocks, 256 chunks), 256 threads (8 waves as 4(M) x 2(N)).
// ---------------------------------------------------------------------------
__global__ __launch_bounds__(256) void proj_gemm_kernel(
    const float* __restrict__ x, const __bf16* __restrict__ wpack,
    __bf16* __restrict__ Hout)
{
    const int chunk = blockIdx.y;
    const int nbase = blockIdx.x * 64;
    const int t = threadIdx.x;
    const int lane = t & 31;
    const int wave = t >> 5;
    const int wm = wave >> 1;   // 0..3
    const int wn = wave & 1;    // 0..1
    const float* xc = x + (size_t)chunk * LC * MOTIF;

    v8f acc[4][2];
    v8f zero = {};
    for (int a = 0; a < 4; ++a)
        for (int nt = 0; nt < 2; ++nt) acc[a][nt] = zero;

    const int khalf = lane >> 4;
    for (int kt = 0; kt < KPROJ / 32; ++kt) {
        BFrag bf[2];
        int kb = kt * 32 + khalf * 16;
        for (int nt = 0; nt < 2; ++nt) {
            int pos = nbase + wn * 32 + nt * 16 + (lane & 15);
            const float* px = xc + (size_t)pos * MOTIF;
#pragma unroll
            for (int j = 0; j < 16; ++j) {
                int k = kb + j;
                float v = (k < MOTIF) ? px[k] : 0.0f;
                bf[nt].e[j] = (__bf16)v;
            }
        }
#pragma unroll
        for (int a = 0; a < 4; ++a) {
            int mt = wm * 4 + a;
            const __bf16* pa = wpack + ((size_t)(mt * 9 + kt) * 32 + lane) * 16;
            BFrag af;
            af.u[0] = *(const uint4*)(pa);
            af.u[1] = *(const uint4*)(pa + 8);
#pragma unroll
            for (int nt = 0; nt < 2; ++nt)
                acc[a][nt] = __builtin_amdgcn_wmma_f32_16x16x32_bf16(
                    false, af.v, false, bf[nt].v, (short)0, acc[a][nt], false, false);
        }
    }

    __bf16* Ho = Hout + (size_t)chunk * LC * HID;
#pragma unroll
    for (int a = 0; a < 4; ++a) {
#pragma unroll
        for (int nt = 0; nt < 2; ++nt) {
            int pos   = nbase + wn * 32 + nt * 16 + (lane & 15);
            int obase = (wm * 4 + a) * 16 + (lane >> 4) * 8;
            union { uint4 u; __bf16 e[8]; } st;
#pragma unroll
            for (int r = 0; r < 8; ++r) st.e[r] = (__bf16)acc[a][nt][r];
            *(uint4*)(Ho + (size_t)pos * HID + obase) = st.u;
        }
    }
}

// ---------------------------------------------------------------------------
// Dilated conv layer as implicit GEMM, M=256, N=64/block, K=768.
// Per K-step (one tap, 32 in-channels) a 64x32 bf16 activation slab is staged
// into a double-buffered LDS tile — via async-to-LDS when available — while
// the previous slab feeds 8 WMMAs per wave.
// Epilogue: c += bias; h = gelu(c) + c; store bf16.
// ---------------------------------------------------------------------------
__global__ __launch_bounds__(256) void conv_tower_kernel(
    const __bf16* __restrict__ Hin, __bf16* __restrict__ Hout,
    const __bf16* __restrict__ wpack, const float* __restrict__ bias,
    int layer, int dil)
{
    __shared__ __align__(16) __bf16 s[2][64 * 32];   // [buf][pos 64][ci 32]
    const int chunk = blockIdx.y;
    const int nbase = blockIdx.x * 64;
    const int t = threadIdx.x;
    const int lane = t & 31;
    const int wave = t >> 5;
    const int wm = wave >> 1;
    const int wn = wave & 1;
    const __bf16* Hc = Hin + (size_t)chunk * LC * HID;
    __bf16* Ho = Hout + (size_t)chunk * LC * HID;
    const __bf16* wl = wpack + (size_t)layer * 16 * 24 * 512;

    float biasv[4][8];
#pragma unroll
    for (int a = 0; a < 4; ++a)
#pragma unroll
        for (int r = 0; r < 8; ++r) {
            int o = (wm * 4 + a) * 16 + (lane >> 4) * 8 + r;
            biasv[a][r] = bias[layer * HID + o];
        }

    v8f acc[4][2];
    v8f zero = {};
    for (int a = 0; a < 4; ++a)
        for (int nt = 0; nt < 2; ++nt) acc[a][nt] = zero;

    const int stage_pos = t >> 2;   // 0..63
    const int stage_q   = t & 3;    // 0..3 (8 channels each)
    const int khalf = lane >> 4;

    // stage K-step kt's activation slab into s[kt&1]
    auto stage = [&](int kt) {
        int tap    = kt >> 3;
        int shift  = dil * (tap - 1);
        int cibase = (kt & 7) * 32;
        int pg = nbase + stage_pos + shift;
        __bf16* dst = &s[kt & 1][stage_pos * 32 + stage_q * 8];
        if (pg >= 0 && pg < LC) {
            const __bf16* src = Hc + (size_t)pg * HID + cibase + stage_q * 8;
#if defined(USE_ASYNC_LDS)
            __builtin_amdgcn_global_load_async_to_lds_b128(
                (__attribute__((address_space(1))) v4i_vec*)src,
                (__attribute__((address_space(3))) v4i_vec*)dst,
                0, 0);
#else
            *(uint4*)dst = *(const uint4*)src;
#endif
        } else {
            *(uint4*)dst = make_uint4(0u, 0u, 0u, 0u);
        }
    };

    stage(0);
    wait_async_all();
    __syncthreads();

    for (int kt = 0; kt < 24; ++kt) {
        if (kt < 23) stage(kt + 1);
        const __bf16* sb = &s[kt & 1][0];
        // ---- B fragments from LDS (ds_load_b128 x2 per fragment) ----
        BFrag bf[2];
#pragma unroll
        for (int nt = 0; nt < 2; ++nt) {
            int pl = wn * 32 + nt * 16 + (lane & 15);
            const __bf16* p = sb + pl * 32 + khalf * 16;
            bf[nt].u[0] = *(const uint4*)(p);
            bf[nt].u[1] = *(const uint4*)(p + 8);
        }
        // ---- A fragments from packed weights (L2) + WMMA ----
#pragma unroll
        for (int a = 0; a < 4; ++a) {
            int mt = wm * 4 + a;
            const __bf16* pa = wl + ((size_t)(mt * 24 + kt) * 32 + lane) * 16;
            if (kt < 23)
                __builtin_prefetch(wl + ((size_t)(mt * 24 + kt + 1) * 32 + lane) * 16, 0, 0);
            BFrag af;
            af.u[0] = *(const uint4*)(pa);
            af.u[1] = *(const uint4*)(pa + 8);
#pragma unroll
            for (int nt = 0; nt < 2; ++nt)
                acc[a][nt] = __builtin_amdgcn_wmma_f32_16x16x32_bf16(
                    false, af.v, false, bf[nt].v, (short)0, acc[a][nt], false, false);
        }
        wait_async_all();
        __syncthreads();
    }

    // ---- epilogue: bias + exact GELU residual, bf16 store ----
#pragma unroll
    for (int a = 0; a < 4; ++a) {
#pragma unroll
        for (int nt = 0; nt < 2; ++nt) {
            int pos   = nbase + wn * 32 + nt * 16 + (lane & 15);
            int obase = (wm * 4 + a) * 16 + (lane >> 4) * 8;
            union { uint4 u; __bf16 e[8]; } st;
#pragma unroll
            for (int r = 0; r < 8; ++r) {
                float c = acc[a][nt][r] + biasv[a][r];
                float h = 0.5f * c * (1.0f + erff(c * 0.70710678118654752f)) + c;
                st.e[r] = (__bf16)h;
            }
            *(uint4*)(Ho + (size_t)pos * HID + obase) = st.u;
        }
    }
}

// ---------------------------------------------------------------------------
// Heads: profile conv (K=20, pad 9/10) + mean-pool + atpm with n_peaks mask.
// out[0..255] = peak_atpms, out[256..] = peak_profiles (flat, f32).
// ---------------------------------------------------------------------------
__global__ __launch_bounds__(256) void heads_kernel(
    const __bf16* __restrict__ H, const float* __restrict__ w_prof,
    const float* __restrict__ b_prof, const float* __restrict__ w_atpm,
    const float* __restrict__ b_atpm, const long long* __restrict__ n_peaks,
    float* __restrict__ out)
{
    __shared__ float wp[20 * HID];   // transposed [k][ch]
    __shared__ float wa[HID];
    __shared__ float red[256];
    const int chunk = blockIdx.x;
    const int t = threadIdx.x;
    for (int idx = t; idx < 20 * HID; idx += 256) {
        int ch = idx / 20, k = idx % 20;
        wp[k * HID + ch] = w_prof[idx];
    }
    wa[t] = w_atpm[t];
    __syncthreads();

    const __bf16* Hc = H + (size_t)chunk * LC * HID;
    float partial = 0.0f;
#pragma unroll
    for (int li = 0; li < 2; ++li) {
        int l = 2 * t + li;
        float accp = b_prof[0];
        for (int k = 0; k < 20; ++k) {
            int pg = l + k - 9;
            if (pg < 0 || pg >= LC) continue;
            const __bf16* hr = Hc + (size_t)pg * HID;
            const float* wr = &wp[k * HID];
            for (int ch = 0; ch < HID; ch += 8) {
                union { uint4 u; __bf16 e[8]; } hv;
                hv.u = *(const uint4*)(hr + ch);
#pragma unroll
                for (int r = 0; r < 8; ++r) accp += (float)hv.e[r] * wr[ch + r];
            }
        }
        out[256 + (size_t)chunk * LC + l] = accp;
        const __bf16* hl = Hc + (size_t)l * HID;
        float sa = 0.0f;
        for (int ch = 0; ch < HID; ch += 8) {
            union { uint4 u; __bf16 e[8]; } hv;
            hv.u = *(const uint4*)(hl + ch);
#pragma unroll
            for (int r = 0; r < 8; ++r) sa += (float)hv.e[r] * wa[ch + r];
        }
        partial += sa;
    }
    red[t] = partial;
    __syncthreads();
    for (int sft = 128; sft > 0; sft >>= 1) {
        if (t < sft) red[t] += red[t + sft];
        __syncthreads();
    }
    if (t == 0) {
        float atpm = red[0] * (1.0f / (float)LC) + b_atpm[0];
        int b = chunk >> 7, p = chunk & 127;
        out[chunk] = (p < (int)n_peaks[b]) ? atpm : 0.0f;
    }
}

// ---------------------------------------------------------------------------
extern "C" void kernel_launch(void* const* d_in, const int* in_sizes, int n_in,
                              void* d_out, int out_size, void* d_ws, size_t ws_size,
                              hipStream_t stream) {
    (void)in_sizes; (void)n_in; (void)out_size; (void)ws_size;
    const float* x        = (const float*)d_in[0];
    const float* w_proj   = (const float*)d_in[1];
    const float* tower_w  = (const float*)d_in[2];
    const float* tower_b  = (const float*)d_in[3];
    const float* w_prof   = (const float*)d_in[4];
    const float* b_prof   = (const float*)d_in[5];
    const float* w_atpm   = (const float*)d_in[6];
    const float* b_atpm   = (const float*)d_in[7];
    const long long* n_peaks = (const long long*)d_in[8];
    float* out = (float*)d_out;

    char* ws = (char*)d_ws;
    size_t offProj = 0;
    size_t szProj  = (size_t)16 * 9 * 512 * 2;            // 147456 B
    size_t offTow  = (offProj + szProj + 255) & ~(size_t)255;
    size_t szTow   = (size_t)7 * 16 * 24 * 512 * 2;       // ~2.75 MB
    size_t offH0   = (offTow + szTow + 255) & ~(size_t)255;
    size_t szH     = (size_t)NCHUNK * LC * HID * 2;       // 64 MB (bf16)
    size_t offH1   = offH0 + szH;

    __bf16* packProj = (__bf16*)(ws + offProj);
    __bf16* packTow  = (__bf16*)(ws + offTow);
    __bf16* H0       = (__bf16*)(ws + offH0);
    __bf16* H1       = (__bf16*)(ws + offH1);
    __bf16* bufs[2]  = { H0, H1 };

    pack_proj_kernel<<<(16 * 9 * 512 + 255) / 256, 256, 0, stream>>>(w_proj, packProj);
    pack_tower_kernel<<<(7 * 16 * 24 * 512 + 255) / 256, 256, 0, stream>>>(tower_w, packTow);

    proj_gemm_kernel<<<dim3(8, NCHUNK), 256, 0, stream>>>(x, packProj, H0);

    for (int l = 0; l < 7; ++l) {
        conv_tower_kernel<<<dim3(8, NCHUNK), 256, 0, stream>>>(
            bufs[l & 1], bufs[(l + 1) & 1], packTow, tower_b, l, 2 << l);
    }

    heads_kernel<<<NCHUNK, 256, 0, stream>>>(
        bufs[1], w_prof, b_prof, w_atpm, b_atpm, n_peaks, out);
}
